// xGRU_420906795589
// MI455X (gfx1250) — matrix operand order repaired and checked by reading, not compiled, not run on hardware
//
#include <hip/hip_runtime.h>
#include <hip/hip_bf16.h>

// ---------------------------------------------------------------------------
// xGRU (mGRU + sGRU) persistent-scan kernel for MI455X (gfx1250, wave32, WMMA)
//   grid = 8 workgroups (one batch row per WGP), 512 threads = 16 wave32
//   c-state (128KB) lives in LDS (320KB/WGP); weights converted to f16 in d_ws
//   all GEMVs via v_wmma_f32_16x16x32_f16 with replicated-B columns
// ---------------------------------------------------------------------------

typedef __attribute__((ext_vector_type(8)))  _Float16 v8h;
typedef __attribute__((ext_vector_type(16))) _Float16 v16h;
typedef __attribute__((ext_vector_type(8)))  float    v8f;

#define EPS_F 1e-5f
#define NTH 512
#define NWAVE 16
#define SEQ_S 512
#define DIM_H 512
#define DIM_IN 512

// ---- f16 weight arena layout inside d_ws (element offsets, all 32B aligned)
static constexpr size_t OFF_PROJ = 0;         // (512,512)
static constexpr size_t OFF_UPL  = 262144;    // (1024,512)
static constexpr size_t OFF_UPR  = 786432;    // (512,512)
static constexpr size_t OFF_DOWN = 1048576;   // (512,512)
static constexpr size_t OFF_SKIP = 1310720;   // (512,1024)
static constexpr size_t OFF_WI   = 1835008;   // (8->16,1024)
static constexpr size_t OFF_WO   = 1851392;   // (512,1024)
static constexpr size_t OFF_WQ   = 2375680;   // (512,1024)
static constexpr size_t OFF_WK   = 2899968;   // (512,1024)
static constexpr size_t OFF_WV   = 3424256;   // (512,1024)
static constexpr size_t OFF_SW   = 3948544;   // (512,512)
static constexpr size_t OFF_SR   = 4210688;   // (8*64,64)
static constexpr size_t OFF_SUP  = 4243456;   // (682->768,512)  48 chunks = 3/wave
static constexpr size_t OFF_SDN  = 4636672;   // (512,682->704)
static constexpr size_t WS_F16_TOTAL = 4997120;

// ---- LDS layout (float region then f16 region)
static constexpr int F_C    = 0;        // c state, 512 rows x stride 65 (bank-conflict pad)
static constexpr int F_N    = 33280;    // n state (512)
static constexpr int F_SN   = 33792;    // sGRU n
static constexpr int F_SH   = 34304;    // sGRU h
static constexpr int F_SM   = 34816;    // sGRU m
static constexpr int F_XT   = 35328;
static constexpr int F_LEFT = 35840;    // 1024
static constexpr int F_RGT  = 36864;
static constexpr int F_Q    = 37376;
static constexpr int F_K    = 37888;
static constexpr int F_V    = 38400;
static constexpr int F_O    = 38912;
static constexpr int F_SKP  = 39424;
static constexpr int F_NUM  = 39936;    // num, then reused as h-vector
static constexpr int F_Y1   = 40448;
static constexpr int F_WX   = 40960;
static constexpr int F_BLIN = 41472;
static constexpr int F_IEA  = 41984;
static constexpr int F_FEA  = 42496;
static constexpr int F_YUP  = 43008;    // 768
static constexpr int F_Y2   = 43776;
static constexpr int F_I16  = 44288;    // 16
static constexpr int F_RED  = 44304;    // 48 (reduction scratch)
static constexpr int F_SC   = 44352;    // 48 small per-head scratch
static constexpr int F_TOTAL= 44400;
static constexpr int H_BASE_BYTES = F_TOTAL * 4;      // 177600, 32B aligned
static constexpr int H_SEQ  = 0;
static constexpr int H_XL   = 512;
static constexpr int H_LEFT = 1024;     // 1024
static constexpr int H_XC   = 2048;     // 1024
static constexpr int H_GNO  = 3072;
static constexpr int H_XL2  = 3584;
static constexpr int H_SH   = 4096;
static constexpr int H_IEH  = 4608;
static constexpr int H_GN2  = 5120;
static constexpr int H_OUT3 = 5632;     // 704
static constexpr int H_TOTAL= 6336;
static constexpr int LDS_BYTES = H_BASE_BYTES + H_TOTAL * 2;   // 190272 < 320KB

union AFrag { v16h v; v8h h[2]; };

struct Params {
  const float* seq;
  const _Float16* wf;
  const float *m_ln_g,*m_ln_b,*m_gn_g,*m_gn_b;
  const float *upL_b,*upR_b,*down_b,*conv_w,*conv_b;
  const float *Wi_b,*Wo_b,*Wq_b,*Wk_b,*Wv_b;
  const float *s_ln_g,*s_ln_b,*s_gn_g,*s_gn_b,*sW_b,*sup_b,*sdn_b;
  float* out;
};

// ---------------------------------------------------------------------------
// GEMV via WMMA: y[NC*256] = W[(NC*256) x K] * x[K]; W f16 row-major, x f16 in
// LDS.  B fragment = x chunk replicated into all 16 columns -> D replicated;
// 512 useful MACs per v_wmma_f32_16x16x32_f16.  A and B both use contiguous
// k = k0 + 16*hlf + s.  NC (chunks per wave) is a template constant and the wave index is
// scalarized via readfirstlane, so there is no divergent control flow around
// the WMMAs (EXEC stays all-ones) and no runtime guards.
// ---------------------------------------------------------------------------
template<int NC>
__device__ __forceinline__ void gemv_wmma(const _Float16* __restrict__ W, int K,
                                          const _Float16* __restrict__ xh,
                                          float* __restrict__ y, int tid)
{
  const int lane  = tid & 31;
  const int waveS = __builtin_amdgcn_readfirstlane(tid >> 5);   // SGPR
  const int hlf = lane >> 4, m = lane & 15;
  const size_t rowOff = (size_t)m * (size_t)K;

  const v8f vz = {};
  v8f acc[NC];
  #pragma unroll
  for (int i = 0; i < NC; ++i) acc[i] = vz;

  for (int k0 = 0; k0 < K; k0 += 32) {
    v16h bfrag = *(const v16h*)(xh + k0 + 16 * hlf);   // B: all 16 cols equal
    #pragma unroll
    for (int cc = 0; cc < NC; ++cc) {
      const _Float16* Wrow = W + (size_t)((waveS + cc * NWAVE) << 4) * (size_t)K + rowOff;
      AFrag a;
      a.v = *(const v16h*)(Wrow + k0 + 16 * hlf);
      acc[cc] = __builtin_amdgcn_wmma_f32_16x16x32_f16(false, a.v, false, bfrag,
                                                       (short)0, acc[cc], false, false);
    }
  }

  if (m == 0) {
    #pragma unroll
    for (int cc = 0; cc < NC; ++cc) {
      const int j0 = (waveS + cc * NWAVE) << 4;
      #pragma unroll
      for (int g = 0; g < 8; ++g) y[j0 + 8 * hlf + g] = acc[cc][g];
    }
  }
}

// Nout=16 (Wi): single chunk, wave 0 only (scalar branch, no exec juggling)
__device__ __forceinline__ void gemv_wi(const _Float16* __restrict__ W,
                                        const _Float16* __restrict__ xh,
                                        float* __restrict__ y, int tid)
{
  const int lane  = tid & 31;
  const int waveS = __builtin_amdgcn_readfirstlane(tid >> 5);
  if (waveS != 0) return;
  const int hlf = lane >> 4, m = lane & 15;
  const _Float16* Wrow = W + (size_t)m * 1024;
  v8f acc = {};
  for (int k0 = 0; k0 < 1024; k0 += 32) {
    AFrag a;
    a.v = *(const v16h*)(Wrow + k0 + 16 * hlf);
    v16h bfrag = *(const v16h*)(xh + k0 + 16 * hlf);
    acc = __builtin_amdgcn_wmma_f32_16x16x32_f16(false, a.v, false, bfrag,
                                                 (short)0, acc, false, false);
  }
  if (m == 0) {
    #pragma unroll
    for (int g = 0; g < 8; ++g) y[8 * hlf + g] = acc[g];
  }
}

// Block-diagonal (8 heads x 64x64) for sGRU blin: 32 chunks, 2/wave, K=64
__device__ __forceinline__ void gemv_blockdiag(const _Float16* __restrict__ R,
                                               const _Float16* __restrict__ xh,
                                               float* __restrict__ y, int tid)
{
  const int lane  = tid & 31;
  const int waveS = __builtin_amdgcn_readfirstlane(tid >> 5);
  const int hlf = lane >> 4, m = lane & 15;
  #pragma unroll
  for (int cc = 0; cc < 2; ++cc) {
    const int chunk = waveS + cc * NWAVE;     // scalar
    const int hd = chunk >> 2;
    const int j0 = (chunk & 3) << 4;
    const _Float16* Wrow = R + hd * 4096 + (j0 + m) * 64;
    const _Float16* xb = xh + hd * 64;
    v8f acc = {};
    #pragma unroll
    for (int k0 = 0; k0 < 64; k0 += 32) {
      AFrag a;
      a.v = *(const v16h*)(Wrow + k0 + 16 * hlf);
      v16h bfrag = *(const v16h*)(xb + k0 + 16 * hlf);
      acc = __builtin_amdgcn_wmma_f32_16x16x32_f16(false, a.v, false, bfrag,
                                                   (short)0, acc, false, false);
    }
    if (m == 0) {
      #pragma unroll
      for (int g = 0; g < 8; ++g) y[hd * 64 + j0 + 8 * hlf + g] = acc[g];
    }
  }
}

// two-value block reduction (sum) over 512 threads, wave32 shuffles
__device__ __forceinline__ void block_sum2(float a, float b, float* red, int tid,
                                           float& outa, float& outb)
{
  #pragma unroll
  for (int off = 16; off > 0; off >>= 1) {
    a += __shfl_xor(a, off, 32);
    b += __shfl_xor(b, off, 32);
  }
  if ((tid & 31) == 0) { red[tid >> 5] = a; red[16 + (tid >> 5)] = b; }
  __syncthreads();
  if (tid == 0) {
    float sa = 0.f, sb = 0.f;
    for (int i = 0; i < 16; ++i) { sa += red[i]; sb += red[16 + i]; }
    red[32] = sa; red[33] = sb;
  }
  __syncthreads();
  outa = red[32]; outb = red[33];
}

__device__ __forceinline__ float sigmoidf_(float x) { return 1.f / (1.f + expf(-x)); }

__global__ void __launch_bounds__(NTH)
xgru_scan(Params p)
{
  extern __shared__ char smem[];
  float* F = reinterpret_cast<float*>(smem);
  _Float16* Hh = reinterpret_cast<_Float16*>(smem + H_BASE_BYTES);

  float* c_s  = F + F_C;   float* n_s  = F + F_N;
  float* s_n  = F + F_SN;  float* s_h  = F + F_SH;  float* s_m = F + F_SM;
  float* xt   = F + F_XT;  float* left = F + F_LEFT; float* rgt = F + F_RGT;
  float* qa   = F + F_Q;   float* ka   = F + F_K;   float* va  = F + F_V;
  float* oa   = F + F_O;   float* skp  = F + F_SKP; float* num = F + F_NUM;
  float* y1   = F + F_Y1;  float* Wx   = F + F_WX;  float* blin= F + F_BLIN;
  float* ieA  = F + F_IEA; float* feA  = F + F_FEA; float* yup = F + F_YUP;
  float* y2   = F + F_Y2;  float* i16  = F + F_I16; float* red = F + F_RED;
  float* sc   = F + F_SC;  // m8|ie8|fe8|den8|mu8|var8 (8 each)

  _Float16* seq_h = Hh + H_SEQ;  _Float16* xl_h  = Hh + H_XL;
  _Float16* left_h= Hh + H_LEFT; _Float16* xc_h  = Hh + H_XC;
  _Float16* gno_h = Hh + H_GNO;  _Float16* xl2_h = Hh + H_XL2;
  _Float16* sh_h  = Hh + H_SH;   _Float16* ieh_h = Hh + H_IEH;
  _Float16* gn2_h = Hh + H_GN2;  _Float16* out3_h= Hh + H_OUT3;

  const int tid = threadIdx.x;
  const int bb  = blockIdx.x;
  const _Float16* wf = p.wf;

  // ---- init carried state
  for (int i = tid; i < 512 * 65; i += NTH) c_s[i] = 0.f;
  n_s[tid] = 1.f;                 // n init ones
  s_n[tid] = 1.f; s_h[tid] = 0.f; s_m[tid] = 0.f;
  sh_h[tid] = (_Float16)0.f;
  if (tid < 8) sc[tid] = 0.f;     // m8 = 0
  __syncthreads();

  const float cw0 = p.conv_w[0], cw1 = p.conv_w[1], cw2 = p.conv_w[2], cw3 = p.conv_w[3];
  const float cvb = p.conv_b[0];
  const float inv_sqrt_d = 0.125f;           // 1/sqrt(64)
  const float E_CONST = 2.718281828459045f;

  for (int t = 0; t < SEQ_S; ++t) {
    const float* srow = p.seq + ((size_t)bb * SEQ_S + t) * DIM_IN;
    seq_h[tid] = (_Float16)srow[tid];
    if (t + 1 < SEQ_S) __builtin_prefetch(srow + DIM_IN + tid, 0, 0); // next-row warm (global_prefetch_b8)
    __syncthreads();

    // xt = seq[t] @ proj_W^T
    gemv_wmma<2>(wf + OFF_PROJ, 512, seq_h, xt, tid);
    __syncthreads();

    // ---- mGRU step ------------------------------------------------------
    { // LayerNorm
      float v = xt[tid], su, sq;
      block_sum2(v, v * v, red, tid, su, sq);
      float mu = su * (1.f / 512.f);
      float var = sq * (1.f / 512.f) - mu * mu;
      xl_h[tid] = (_Float16)((v - mu) * rsqrtf(var + EPS_F) * p.m_ln_g[tid] + p.m_ln_b[tid]);
    }
    __syncthreads();

    gemv_wmma<4>(wf + OFF_UPL, 512, xl_h, left, tid);
    gemv_wmma<2>(wf + OFF_UPR, 512, xl_h, rgt, tid);
    __syncthreads();

    for (int j = tid; j < 1024; j += NTH) {
      float lv = left[j] + p.upL_b[j];
      left[j] = lv; left_h[j] = (_Float16)lv;
    }
    rgt[tid] += p.upR_b[tid];
    __syncthreads();

    // causal conv (feature axis, left-pad 3) + SiLU -> xc
    for (int j = tid; j < 1024; j += NTH) {
      float s = cvb;
      if (j >= 3) s += cw0 * left[j - 3];
      if (j >= 2) s += cw1 * left[j - 2];
      if (j >= 1) s += cw2 * left[j - 1];
      s += cw3 * left[j];
      xc_h[j] = (_Float16)(s * sigmoidf_(s));
    }
    __syncthreads();

    gemv_wmma<2>(wf + OFF_WQ,   1024, xc_h,   qa,  tid);
    gemv_wmma<2>(wf + OFF_WK,   1024, xc_h,   ka,  tid);
    gemv_wmma<2>(wf + OFF_WV,   1024, left_h, va,  tid);
    gemv_wmma<2>(wf + OFF_WO,   1024, left_h, oa,  tid);
    gemv_wmma<2>(wf + OFF_SKIP, 1024, xc_h,   skp, tid);
    gemv_wi(wf + OFF_WI, xc_h, i16, tid);
    __syncthreads();

    qa[tid] += p.Wq_b[tid];
    ka[tid]  = (ka[tid] + p.Wk_b[tid]) * inv_sqrt_d;
    va[tid] += p.Wv_b[tid];
    oa[tid]  = sigmoidf_(oa[tid] + p.Wo_b[tid]);
    if (tid < 8) {
      float iv = i16[tid] + p.Wi_b[tid];
      float f  = 1.f - iv;
      float mo = sc[tid];
      float mn = fmaxf(f + mo, iv);
      sc[tid]      = mn;                 // m state
      sc[8 + tid]  = expf(iv - mn);      // ie
      sc[16 + tid] = expf(f - mn + mo);  // fe
    }
    __syncthreads();

    { // c/n state update + num (thread = (head h, row d))
      const int hd = tid >> 6;
      const float fe = sc[16 + hd], ie = sc[8 + hd];
      const float vh = va[tid];
      float* crow = c_s + (size_t)tid * 65;
      const float* kh = ka + hd * 64;
      const float* qh = qa + hd * 64;
      float acc = 0.f;
      #pragma unroll 4
      for (int pp = 0; pp < 64; ++pp) {
        float cv = fe * crow[pp] + ie * vh * kh[pp];
        crow[pp] = cv;
        acc += cv * qh[pp];
      }
      num[tid] = acc;
      n_s[tid] = fe * n_s[tid] + ie * ka[tid];
    }
    __syncthreads();
    if (tid < 8) {
      float dn = 0.f;
      for (int pp = 0; pp < 64; ++pp) dn += n_s[tid * 64 + pp] * qa[tid * 64 + pp];
      sc[24 + tid] = fmaxf(dn, 1.f);     // den
    }
    __syncthreads();
    num[tid] = oa[tid] * num[tid] / sc[24 + (tid >> 6)];   // num now holds h-vector
    __syncthreads();

    if (tid < 8) { // per-head GN stats
      float s = 0.f, s2 = 0.f;
      for (int d = 0; d < 64; ++d) { float x = num[tid * 64 + d]; s += x; s2 += x * x; }
      float mu = s * (1.f / 64.f);
      sc[32 + tid] = mu;
      sc[40 + tid] = s2 * (1.f / 64.f) - mu * mu;
    }
    __syncthreads();
    {
      const int hd = tid >> 6;
      float xg = (num[tid] - sc[32 + hd]) * rsqrtf(sc[40 + hd] + EPS_F)
                 * p.m_gn_g[tid] + p.m_gn_b[tid];
      float ov = xg + skp[tid];
      float r  = rgt[tid];
      gno_h[tid] = (_Float16)(ov * (r * sigmoidf_(r)));
    }
    __syncthreads();

    gemv_wmma<2>(wf + OFF_DOWN, 512, gno_h, y1, tid);
    __syncthreads();
    y1[tid] += p.down_b[tid] + xt[tid];    // residual
    __syncthreads();

    // ---- sGRU step ------------------------------------------------------
    {
      float v = y1[tid], su, sq;
      block_sum2(v, v * v, red, tid, su, sq);
      float mu = su * (1.f / 512.f);
      float var = sq * (1.f / 512.f) - mu * mu;
      xl2_h[tid] = (_Float16)((v - mu) * rsqrtf(var + EPS_F) * p.s_ln_g[tid] + p.s_ln_b[tid]);
    }
    __syncthreads();

    gemv_wmma<2>(wf + OFF_SW, 512, xl2_h, Wx, tid);
    gemv_blockdiag(wf + OFF_SR, sh_h, blin, tid);    // blin(h_old)
    __syncthreads();

    {
      float wx = Wx[tid] + p.sW_b[tid];
      Wx[tid] = wx;
      float g  = wx + blin[tid];
      float mo = s_m[tid];
      float mn = fmaxf(g + mo, g);
      s_m[tid] = mn;
      float ie = expf(g - mn), fe = expf(g - mn + mo);
      ieA[tid] = ie; feA[tid] = fe;
      ieh_h[tid] = (_Float16)(ie * s_h[tid]);
    }
    __syncthreads();

    gemv_blockdiag(wf + OFF_SR, ieh_h, blin, tid);   // blin(ie*h_old)
    __syncthreads();

    {
      float z  = tanhf(Wx[tid] + blin[tid]);
      float fe = feA[tid], ie = ieA[tid];
      s_n[tid] = fe * s_n[tid] + ie;
      float hn = (E_CONST - fe) * s_h[tid] + fe * z;
      s_h[tid] = hn;
      sh_h[tid] = (_Float16)hn;
    }
    __syncthreads();

    if (tid < 8) {
      float s = 0.f, s2 = 0.f;
      for (int d = 0; d < 64; ++d) { float x = s_h[tid * 64 + d]; s += x; s2 += x * x; }
      float mu = s * (1.f / 64.f);
      sc[32 + tid] = mu;
      sc[40 + tid] = s2 * (1.f / 64.f) - mu * mu;
    }
    __syncthreads();
    {
      const int hd = tid >> 6;
      gn2_h[tid] = (_Float16)((s_h[tid] - sc[32 + hd]) * rsqrtf(sc[40 + hd] + EPS_F)
                              * p.s_gn_g[tid] + p.s_gn_b[tid]);
    }
    __syncthreads();

    gemv_wmma<3>(wf + OFF_SUP, 512, gn2_h, yup, tid);   // 768 rows (682 real + zero pad)
    __syncthreads();
    for (int j = tid; j < 704; j += NTH) {
      float val = 0.f;
      if (j < 682) {
        float yv = yup[j] + p.sup_b[j];
        val = yv + 0.5f * yv * (1.f + erff(yv * 0.70710678118654752f)); // y + gelu_exact(y)
      }
      out3_h[j] = (_Float16)val;
    }
    __syncthreads();

    gemv_wmma<2>(wf + OFF_SDN, 704, out3_h, y2, tid);
    __syncthreads();

    p.out[((size_t)bb * SEQ_S + t) * DIM_H + tid] = y2[tid] + p.sdn_b[tid] + y1[tid];
    __syncthreads();   // protect buffers before next step
  }
}

// f32 -> f16 weight conversion with zero padding to (rp, cp)
__global__ void cvt_pad_f16(const float* __restrict__ src, _Float16* __restrict__ dst,
                            int rows, int cols, int rp, int cp)
{
  int i = blockIdx.x * blockDim.x + threadIdx.x;
  int n = rp * cp;
  if (i >= n) return;
  int r = i / cp, c = i - r * cp;
  float v = (r < rows && c < cols) ? src[(size_t)r * cols + c] : 0.f;
  dst[i] = (_Float16)v;
}

struct CvtJob { int src; size_t off; int r, c, rp, cp; };

extern "C" void kernel_launch(void* const* d_in, const int* in_sizes, int n_in,
                              void* d_out, int out_size, void* d_ws, size_t ws_size,
                              hipStream_t stream) {
  (void)in_sizes; (void)n_in; (void)out_size;
  _Float16* wf = reinterpret_cast<_Float16*>(d_ws);

  static const CvtJob jobs[] = {
    {1,  OFF_PROJ, 512, 512, 512, 512},
    {6,  OFF_UPL, 1024, 512, 1024, 512},
    {8,  OFF_UPR,  512, 512, 512, 512},
    {10, OFF_DOWN, 512, 512, 512, 512},
    {14, OFF_SKIP, 512, 1024, 512, 1024},
    {15, OFF_WI,     8, 1024,  16, 1024},
    {17, OFF_WO,   512, 1024, 512, 1024},
    {19, OFF_WQ,   512, 1024, 512, 1024},
    {21, OFF_WK,   512, 1024, 512, 1024},
    {23, OFF_WV,   512, 1024, 512, 1024},
    {29, OFF_SW,   512, 512, 512, 512},
    {31, OFF_SR,   512,  64, 512,  64},
    {32, OFF_SUP,  682, 512, 768, 512},
    {34, OFF_SDN,  512, 682, 512, 704},
  };
  if (ws_size >= WS_F16_TOTAL * sizeof(_Float16)) {
    for (const CvtJob& j : jobs) {
      int n = j.rp * j.cp;
      cvt_pad_f16<<<(n + 255) / 256, 256, 0, stream>>>(
          reinterpret_cast<const float*>(d_in[j.src]), wf + j.off, j.r, j.c, j.rp, j.cp);
    }
  }

  Params p;
  p.seq    = reinterpret_cast<const float*>(d_in[0]);
  p.wf     = wf;
  p.m_ln_g = (const float*)d_in[2];  p.m_ln_b = (const float*)d_in[3];
  p.m_gn_g = (const float*)d_in[4];  p.m_gn_b = (const float*)d_in[5];
  p.upL_b  = (const float*)d_in[7];  p.upR_b  = (const float*)d_in[9];
  p.down_b = (const float*)d_in[11];
  p.conv_w = (const float*)d_in[12]; p.conv_b = (const float*)d_in[13];
  p.Wi_b   = (const float*)d_in[16]; p.Wo_b   = (const float*)d_in[18];
  p.Wq_b   = (const float*)d_in[20]; p.Wk_b   = (const float*)d_in[22];
  p.Wv_b   = (const float*)d_in[24];
  p.s_ln_g = (const float*)d_in[25]; p.s_ln_b = (const float*)d_in[26];
  p.s_gn_g = (const float*)d_in[27]; p.s_gn_b = (const float*)d_in[28];
  p.sW_b   = (const float*)d_in[30];
  p.sup_b  = (const float*)d_in[33]; p.sdn_b  = (const float*)d_in[35];
  p.out    = reinterpret_cast<float*>(d_out);

  (void)hipFuncSetAttribute(reinterpret_cast<const void*>(xgru_scan),
                            hipFuncAttributeMaxDynamicSharedMemorySize, LDS_BYTES);
  xgru_scan<<<8, NTH, LDS_BYTES, stream>>>(p);
}
